// GroupGAT_24438363914896
// MI455X (gfx1250) — matrix operand order repaired and checked by reading, not compiled, not run on hardware
//
#include <hip/hip_runtime.h>
#include <hip/hip_fp16.h>
#include <stdint.h>

typedef __attribute__((ext_vector_type(16))) _Float16 v16h;
typedef __attribute__((ext_vector_type(8)))  _Float16 v8h;
typedef __attribute__((ext_vector_type(8)))  float    v8f;
typedef __attribute__((ext_vector_type(4)))  float    v4f;

#define B_TOTAL 16384
#define NN 41          // nodes per batch element
#define NG 21          // nodes per group (self + 20)
#define D  128
#define TB 8           // batch items per workgroup
#define ROWS  (TB*NG)  // 168 valid GEMM rows
#define ROWSP 176      // padded to 11 tiles of 16
#define MT    (ROWSP/16)

__device__ __forceinline__ int node_of(int n, int side) {
  // group-local node -> global node index in h / mask
  return (n == 0) ? 0 : (side == 0 ? n : 20 + n);
}

__global__ __launch_bounds__(256) void groupgat_kernel(
    const float* __restrict__ h,
    const float* __restrict__ W_ally,
    const float* __restrict__ W_opp,
    const float* __restrict__ a_ally,
    const float* __restrict__ a_opp,
    const unsigned char* __restrict__ mask,   // jax bool: 1 byte/elem
    float* __restrict__ out)
{
  __shared__ _Float16 sH[ROWSP * D];     // 45 KB: H for current side (f16)
  __shared__ _Float16 sA[2][16 * D];     // 2 x 4 KB: double-buffered A tile (f16)
  __shared__ float    sAvec[2 * D];      // attention vector
  __shared__ float    sE[ROWS];          // logits
  __shared__ float    sAttn[ROWS];       // softmax weights
  __shared__ float    sOut[TB * D];      // H0 + ally_ws accumulator

  const int tid  = threadIdx.x;
  const int lane = tid & 31;
  const int wave = tid >> 5;
  const int b0   = blockIdx.x * TB;

  // A staging assignment: thread -> (row, 8-col block)
  const int arow = tid >> 4;           // 0..15
  const int acol = (tid & 15) << 3;    // 0,8,...,120
  // WMMA fragment lane geometry
  const int l15  = lane & 15;
  const int kbA  = (lane >> 4) << 3;   // A: 0 or 8
  const int kb2  = (lane >> 4) << 4;   // B: 0 or 16
  const int ncol = (wave << 4) + l15;  // this wave's output column

  for (int side = 0; side < 2; ++side) {
    const float* W  = (side == 0) ? W_ally : W_opp;
    const float* av = (side == 0) ? a_ally : a_opp;

    sAvec[tid] = av[tid];  // 256 threads, 256 floats

    // ---- Preload B fragments (W columns) into registers, f32 -> f16 ----
    // 16-bit B 32x16, contiguous k order: lane holds col = lane%16, halves j -> K = kb2 + j
    v16h Bf[4];
    #pragma unroll
    for (int ks = 0; ks < 4; ++ks) {
      const float* wp = W + (ks * 32 + kb2) * D + ncol;
      #pragma unroll
      for (int j = 0; j < 16; ++j)
        Bf[ks][j] = (_Float16)wp[j * D];
    }

    // helper lambda: source pointer for this thread's slice of tile mt
    auto srcOf = [&](int mt) -> const float* {
      int m = mt * 16 + arow;
      if (m >= ROWS) m = ROWS - 1;     // pad rows: valid addr, results ignored
      int lb = m / NG, n = m - lb * NG;
      return h + ((size_t)(b0 + lb) * NN + node_of(n, side)) * D + acol;
    };
    auto stage = [&](int mt, int buf) {
      const float* src = srcOf(mt);
      v4f x0 = *(const v4f*)src;
      v4f x1 = *(const v4f*)(src + 4);
      v8h hh;
      #pragma unroll
      for (int j = 0; j < 4; ++j) { hh[j] = (_Float16)x0[j]; hh[4 + j] = (_Float16)x1[j]; }
      *(v8h*)(&sA[buf][arow * D + acol]) = hh;
    };

    // ---- GEMM: H[row, ncol-tile] = A(rows) x W, single barrier per tile ----
    stage(0, 0);
    for (int mt = 0; mt < MT; ++mt) {
      __syncthreads();   // buf[mt&1] written; buf[(mt+1)&1] readers of prev round done
      const int cur = mt & 1;
      if (mt + 2 < MT) __builtin_prefetch(srcOf(mt + 2), 0, 1);  // global_prefetch_b8
      if (mt + 1 < MT) stage(mt + 1, cur ^ 1);

      // A fragments: 16-bit A 16x32, contiguous k order: lane row = lane%16,
      // halves j -> K = kb2 + j (matches B)
      v16h Af[4];
      #pragma unroll
      for (int ks = 0; ks < 4; ++ks) {
        v8h lo = *(const v8h*)(&sA[cur][l15 * D + ks * 32 + kb2]);
        v8h hi = *(const v8h*)(&sA[cur][l15 * D + ks * 32 + kb2 + 8]);
        Af[ks] = __builtin_shufflevector(lo, hi, 0,1,2,3,4,5,6,7,8,9,10,11,12,13,14,15);
      }

      v8f C = {};
      #pragma unroll
      for (int ks = 0; ks < 4; ++ks)
        C = __builtin_amdgcn_wmma_f32_16x16x32_f16(
              /*neg_a=*/false, Af[ks], /*neg_b=*/false, Bf[ks],
              /*c_mod=*/(short)0, C, /*reuse_a=*/false, /*reuse_b=*/false);

      // C layout: VGPR i, lanes0-15 -> M=i, lanes16-31 -> M=8+i; N = lane%16
      {
        int mb = mt * 16 + ((lane < 16) ? 0 : 8);
        #pragma unroll
        for (int i = 0; i < 8; ++i)
          sH[(mb + i) * D + ncol] = (_Float16)C[i];
      }
    }
    __syncthreads();  // all of sH for this side ready

    // ---- logits: e = leaky_relu(H0.a_lo + Hn.a_hi), then mask ----
    for (int p = tid; p < ROWS; p += 256) {
      int lb = p / NG, n = p - lb * NG;
      int base = lb * NG;
      float acc = 0.f;
      #pragma unroll 4
      for (int c = 0; c < D; c += 8) {
        v8h h0 = *(const v8h*)(&sH[ base      * D + c]);
        v8h hn = *(const v8h*)(&sH[(base + n) * D + c]);
        #pragma unroll
        for (int j = 0; j < 8; ++j)
          acc += (float)h0[j] * sAvec[c + j] + (float)hn[j] * sAvec[D + c + j];
      }
      float e = (acc > 0.f) ? acc : 0.2f * acc;           // leaky_relu 0.2
      if (n > 0 && mask[(size_t)(b0 + lb) * NN + node_of(n, side)])
        e = -1.0e9f;                                       // NEG_INF
      sE[p] = e;
    }
    __syncthreads();

    // ---- softmax over 21 nodes per batch item ----
    if (tid < TB) {
      int base = tid * NG;
      float mx = -3.4e38f;
      for (int n = 0; n < NG; ++n) mx = fmaxf(mx, sE[base + n]);
      float s = 0.f;
      for (int n = 0; n < NG; ++n) {
        float ex = __expf(sE[base + n] - mx);  // masked -> exp(-1e9)=0 == where(mask,0,attn)
        sAttn[base + n] = ex; s += ex;
      }
      float inv = 1.f / s;
      for (int n = 0; n < NG; ++n) sAttn[base + n] *= inv;
    }
    __syncthreads();

    // ---- weighted sum; side0 seeds with H_ally[:,0,:], side1 adds + ELU ----
    for (int q = tid; q < TB * D; q += 256) {
      int lb = q >> 7, c = q & (D - 1);
      int base = lb * NG;
      float acc = 0.f;
      #pragma unroll
      for (int n = 0; n < NG; ++n)
        acc += sAttn[base + n] * (float)sH[(base + n) * D + c];
      if (side == 0) {
        sOut[q] = (float)sH[base * D + c] + acc;   // H_ally[:,0,:] + ally_ws
      } else {
        float x = sOut[q] + acc;
        out[(size_t)(b0 + lb) * D + c] = (x > 0.f) ? x : __expf(x) - 1.f;  // elu
      }
    }
    __syncthreads();  // protect sH/sAvec/sE/sAttn reuse by next side
  }
}

extern "C" void kernel_launch(void* const* d_in, const int* in_sizes, int n_in,
                              void* d_out, int out_size, void* d_ws, size_t ws_size,
                              hipStream_t stream) {
  const float*         h      = (const float*)d_in[0];
  const float*         W_ally = (const float*)d_in[1];
  const float*         W_opp  = (const float*)d_in[2];
  const float*         a_ally = (const float*)d_in[3];
  const float*         a_opp  = (const float*)d_in[4];
  const unsigned char* mask   = (const unsigned char*)d_in[5];
  // d_in[6], d_in[7] = num_ally/num_opp scalars (compile-time constants here)
  float* out = (float*)d_out;

  dim3 grid(B_TOTAL / TB);   // 2048 workgroups
  groupgat_kernel<<<grid, 256, 0, stream>>>(h, W_ally, W_opp, a_ally, a_opp, mask, out);

  (void)in_sizes; (void)n_in; (void)out_size; (void)d_ws; (void)ws_size;
}
